// DecoderWithAttention_11742440587337
// MI455X (gfx1250) — matrix operand order repaired and checked by reading, not compiled, not run on hardware
//
#include <hip/hip_runtime.h>
#include <hip/hip_bf16.h>
#include <math.h>

typedef __attribute__((ext_vector_type(16))) __bf16 v16bf;
typedef __attribute__((ext_vector_type(8)))  __bf16 v8bf;
typedef __attribute__((ext_vector_type(8)))  float  v8f;
typedef __attribute__((ext_vector_type(4)))  float  v4f;

#define B_   128
#define P_   196
#define ENC_ 2048
#define EMB_ 512
#define DEC_ 512
#define ATT_ 512
#define V_   10000
#define T_   20
#define XDIM_ (EMB_ + ENC_ + DEC_)   // 3072, K for fused LSTM gate GEMM
#define G4_  (4 * DEC_)              // 2048, N for fused LSTM gate GEMM

// ---------------------------------------------------------------------------
// bf16 WMMA GEMM:  C[M,N] = A_f32[M,K] * B^T_bf16[N,K] + bias[N]
// One wave computes a 16x64 tile: 4 N-subtiles -> 4 independent accumulator
// chains per K-chunk (WMMA ILP, no RAW serialization) and the A fragment is
// loaded + converted once and reused by all 4 v_wmma_f32_16x16x32_bf16.
// Requires M%16==0, K%32==0. BT must be allocated with rows padded to a
// multiple of 64 columns (stores of pad subtiles are skipped wave-uniformly).
// ---------------------------------------------------------------------------
__global__ __launch_bounds__(128)
void gemm_bf16_wmma(const float* __restrict__ A, const __bf16* __restrict__ BT,
                    const float* __restrict__ bias, float* __restrict__ C,
                    int M, int N, int K, int ldc)
{
    const int lane = threadIdx.x & 31;
    const int wid  = threadIdx.x >> 5;
    const int nt   = N >> 4;                  // valid 16-col tiles
    const int ng   = (nt + 3) >> 2;           // groups of 4 tiles
    const int wt   = blockIdx.x * 4 + wid;    // wave-uniform
    if (wt >= (M >> 4) * ng) return;          // whole wave exits -> EXEC stays full
    const int tm  = wt / ng;
    const int tg  = wt % ng;
    const int hl  = lane >> 4;                // lane half (0 or 1)
    const int l16 = lane & 15;

    const float*  arow = A + (size_t)(tm * 16 + l16) * K;
    const __bf16* brow[4];
#pragma unroll
    for (int j = 0; j < 4; ++j)
        brow[j] = BT + (size_t)((tg * 4 + j) * 16 + l16) * K;

    v8f acc[4] = {{}, {}, {}, {}};
    for (int k0 = 0; k0 < K; k0 += 32) {
        if (k0 + 32 < K)                       // stream-ahead: global_prefetch path
            __builtin_prefetch(arow + k0 + 32, 0, 0);

        // A fragment (shared by all 4 subtiles): elems 0-7 -> K=k0+hl*8.. ;
        // elems 8-15 -> K=k0+16+hl*8..
        v4f a0 = *(const v4f*)(arow + k0 + hl * 8);
        v4f a1 = *(const v4f*)(arow + k0 + hl * 8 + 4);
        v4f a2 = *(const v4f*)(arow + k0 + 16 + hl * 8);
        v4f a3 = *(const v4f*)(arow + k0 + 16 + hl * 8 + 4);
        v16bf af;
#pragma unroll
        for (int e = 0; e < 4; ++e) {
            af[e]      = (__bf16)a0[e];
            af[e + 4]  = (__bf16)a1[e];
            af[e + 8]  = (__bf16)a2[e];
            af[e + 12] = (__bf16)a3[e];
        }

#pragma unroll
        for (int j = 0; j < 4; ++j) {
            // B fragment: elems 0-7 -> K=k0+hl*8.. ; elems 8-15 -> K=k0+16+hl*8..
            v8bf b0 = *(const v8bf*)(brow[j] + k0 + hl * 8);
            v8bf b1 = *(const v8bf*)(brow[j] + k0 + 16 + hl * 8);
            v16bf bfv;
#pragma unroll
            for (int e = 0; e < 8; ++e) { bfv[e] = b0[e]; bfv[e + 8] = b1[e]; }
            acc[j] = __builtin_amdgcn_wmma_f32_16x16x32_bf16(
                         false, af, false, bfv, (short)0, acc[j], false, false);
        }
    }

    // C/D layout: lane half selects rows tm*16 + hl*8 + r, col = subtile*16 + l16
    float* crow = C + (size_t)(tm * 16 + hl * 8) * ldc;
#pragma unroll
    for (int j = 0; j < 4; ++j) {
        const int tn = tg * 4 + j;
        if (tn < nt) {                         // wave-uniform pad guard
            const int n  = tn * 16 + l16;
            const float bv = bias ? bias[n] : 0.0f;
#pragma unroll
            for (int r = 0; r < 8; ++r)
                crow[(size_t)r * ldc + n] = acc[j][r] + bv;
        }
    }
}

// ---------------------------------------------------------------------------
// One-time weight conversion: dst_bf16[n*K + k] = (bf16) src_f32[k*N + n]
// ---------------------------------------------------------------------------
__global__ void convT_kernel(const float* __restrict__ src, __bf16* __restrict__ dst,
                             int K, int N)
{
    int id = blockIdx.x * blockDim.x + threadIdx.x;
    if (id >= K * N) return;
    int n = id / K, k = id % K;
    dst[id] = (__bf16)src[(size_t)k * N + n];
}

// Fused [W_ih ; W_hh] -> W_cat^T bf16  (K = 3072 rows, N = 2048 cols)
__global__ void convT_cat_kernel(const float* __restrict__ W_ih,
                                 const float* __restrict__ W_hh,
                                 __bf16* __restrict__ dst)
{
    int id = blockIdx.x * blockDim.x + threadIdx.x;
    if (id >= G4_ * XDIM_) return;
    int n = id / XDIM_, k = id % XDIM_;
    float v = (k < EMB_ + ENC_) ? W_ih[(size_t)k * G4_ + n]
                                : W_hh[(size_t)(k - (EMB_ + ENC_)) * G4_ + n];
    dst[id] = (__bf16)v;
}

__global__ void bias_cat_kernel(const float* __restrict__ b_ih,
                                const float* __restrict__ b_hh,
                                float* __restrict__ bc)
{
    int id = blockIdx.x * blockDim.x + threadIdx.x;
    if (id < G4_) bc[id] = b_ih[id] + b_hh[id];
}

// mean over P axis of enc_out [B,P,ENC]
__global__ void mean_kernel(const float* __restrict__ enc_out, float* __restrict__ mean)
{
    int id = blockIdx.x * blockDim.x + threadIdx.x;
    if (id >= B_ * ENC_) return;
    int b = id / ENC_, e = id % ENC_;
    const float* p = enc_out + (size_t)b * P_ * ENC_ + e;
    float s = 0.f;
    for (int i = 0; i < P_; ++i) s += p[(size_t)i * ENC_];
    mean[id] = s * (1.0f / P_);
}

// scores[b,p] = sum_a relu(enc_proj[b,p,a] + dec_att[b,a]) * w_att[a] + b_att
__global__ __launch_bounds__(128)
void scores_kernel(const float* __restrict__ enc_proj, const float* __restrict__ dec_att,
                   const float* __restrict__ w_att, const float* __restrict__ b_att,
                   float* __restrict__ scores)
{
    int lane = threadIdx.x & 31, wid = threadIdx.x >> 5;
    int row = blockIdx.x * 4 + wid;
    if (row >= B_ * P_) return;
    int b = row / P_;
    const float* ep = enc_proj + (size_t)row * ATT_;
    const float* da = dec_att + (size_t)b * ATT_;
    float acc = 0.f;
    for (int a = lane; a < ATT_; a += 32) {
        float v = ep[a] + da[a];
        acc += (v > 0.f ? v : 0.f) * w_att[a];
    }
#pragma unroll
    for (int off = 16; off; off >>= 1) acc += __shfl_xor(acc, off, 32);
    if (lane == 0) scores[row] = acc + b_att[0];
}

// softmax over P=196, one block per batch element
__global__ __launch_bounds__(256)
void softmax_kernel(const float* __restrict__ scores, float* __restrict__ alpha)
{
    __shared__ float sm[256];
    int b = blockIdx.x, tid = threadIdx.x;
    float v = (tid < P_) ? scores[b * P_ + tid] : -3.4e38f;
    sm[tid] = v; __syncthreads();
    for (int off = 128; off; off >>= 1) {
        if (tid < off) sm[tid] = fmaxf(sm[tid], sm[tid + off]);
        __syncthreads();
    }
    float mx = sm[0]; __syncthreads();
    float e = (tid < P_) ? __expf(v - mx) : 0.f;
    sm[tid] = e; __syncthreads();
    for (int off = 128; off; off >>= 1) {
        if (tid < off) sm[tid] += sm[tid + off];
        __syncthreads();
    }
    float inv = 1.0f / sm[0];
    if (tid < P_) alpha[b * P_ + tid] = e * inv;
}

// context[b,e] = sum_p alpha[b,p] * enc_out[b,p,e]
__global__ void context_kernel(const float* __restrict__ alpha,
                               const float* __restrict__ enc_out,
                               float* __restrict__ context)
{
    int id = blockIdx.x * blockDim.x + threadIdx.x;
    if (id >= B_ * ENC_) return;
    int b = id / ENC_, e = id % ENC_;
    const float* eo = enc_out + (size_t)b * P_ * ENC_ + e;
    const float* al = alpha + b * P_;
    float s = 0.f;
    for (int p = 0; p < P_; ++p) s += al[p] * eo[(size_t)p * ENC_];
    context[id] = s;
}

// x = [ emb[captions[:,t]] , sigmoid(gate_pre)*context , h ]   (fused gather + gate)
__global__ void build_x_kernel(const float* __restrict__ emb, const int* __restrict__ captions,
                               const float* __restrict__ gate_pre, const float* __restrict__ context,
                               const float* __restrict__ h, float* __restrict__ x, int t)
{
    int id = blockIdx.x * blockDim.x + threadIdx.x;
    if (id >= B_ * XDIM_) return;
    int b = id / XDIM_, j = id % XDIM_;
    float v;
    if (j < EMB_) {
        int tok = captions[b * T_ + t];
        v = emb[(size_t)tok * EMB_ + j];
    } else if (j < EMB_ + ENC_) {
        int e = j - EMB_;
        float g = 1.0f / (1.0f + __expf(-gate_pre[b * ENC_ + e]));
        v = g * context[b * ENC_ + e];
    } else {
        v = h[b * DEC_ + (j - (EMB_ + ENC_))];
    }
    x[id] = v;
}

// LSTM cell elementwise: gates layout [i | f | g | o]
__global__ void lstm_kernel(const float* __restrict__ gates,
                            float* __restrict__ h, float* __restrict__ c)
{
    int id = blockIdx.x * blockDim.x + threadIdx.x;
    if (id >= B_ * DEC_) return;
    int b = id / DEC_, d = id % DEC_;
    const float* g = gates + (size_t)b * G4_;
    float i  = 1.0f / (1.0f + __expf(-g[d]));
    float f  = 1.0f / (1.0f + __expf(-g[d + DEC_]));
    float gg = tanhf(g[d + 2 * DEC_]);
    float o  = 1.0f / (1.0f + __expf(-g[d + 3 * DEC_]));
    float cn = f * c[id] + i * gg;
    c[id] = cn;
    h[id] = o * tanhf(cn);
}

// ---------------------------------------------------------------------------
extern "C" void kernel_launch(void* const* d_in, const int* in_sizes, int n_in,
                              void* d_out, int out_size, void* d_ws, size_t ws_size,
                              hipStream_t stream)
{
    (void)in_sizes; (void)n_in; (void)out_size; (void)ws_size;
    const float* enc_out   = (const float*)d_in[0];
    const int*   captions  = (const int*)d_in[1];
    const float* emb       = (const float*)d_in[2];
    const float* W_init_h  = (const float*)d_in[3];
    const float* b_init_h  = (const float*)d_in[4];
    const float* W_init_c  = (const float*)d_in[5];
    const float* b_init_c  = (const float*)d_in[6];
    const float* W_enc_att = (const float*)d_in[7];
    const float* b_enc_att = (const float*)d_in[8];
    const float* W_dec_att = (const float*)d_in[9];
    const float* b_dec_att = (const float*)d_in[10];
    const float* w_att     = (const float*)d_in[11];
    const float* b_att     = (const float*)d_in[12];
    const float* W_fbeta   = (const float*)d_in[13];
    const float* b_fbeta   = (const float*)d_in[14];
    const float* W_ih      = (const float*)d_in[15];
    const float* b_ih      = (const float*)d_in[16];
    const float* W_hh      = (const float*)d_in[17];
    const float* b_hh      = (const float*)d_in[18];
    const float* W_fc      = (const float*)d_in[19];
    const float* b_fc      = (const float*)d_in[20];
    float* out = (float*)d_out;

    const int V_PAD = ((V_ / 16 + 3) / 4) * 4 * 16;   // 10048: pad N to tile-group

    // ---- scratch bump allocator -------------------------------------------
    char* ws = (char*)d_ws;
    size_t off = 0;
    auto alloc = [&](size_t bytes) -> void* {
        void* p = ws + off;
        off += (bytes + 255) & ~(size_t)255;
        return p;
    };
    float*  enc_proj = (float*) alloc((size_t)B_ * P_ * ATT_ * 4);   // 51.4 MB
    __bf16* WencT    = (__bf16*)alloc((size_t)ATT_ * ENC_ * 2);
    __bf16* WdecT    = (__bf16*)alloc((size_t)ATT_ * DEC_ * 2);
    __bf16* WfbT     = (__bf16*)alloc((size_t)ENC_ * DEC_ * 2);
    __bf16* WcatT    = (__bf16*)alloc((size_t)G4_ * XDIM_ * 2);      // 12.6 MB
    __bf16* WfcT     = (__bf16*)alloc((size_t)V_PAD * DEC_ * 2);     // 10.3 MB (padded rows)
    __bf16* WinHT    = (__bf16*)alloc((size_t)DEC_ * ENC_ * 2);
    __bf16* WinCT    = (__bf16*)alloc((size_t)DEC_ * ENC_ * 2);
    float*  bias_cat = (float*) alloc((size_t)G4_ * 4);
    float*  meanb    = (float*) alloc((size_t)B_ * ENC_ * 4);
    float*  h        = (float*) alloc((size_t)B_ * DEC_ * 4);
    float*  c        = (float*) alloc((size_t)B_ * DEC_ * 4);
    float*  dec_att  = (float*) alloc((size_t)B_ * ATT_ * 4);
    float*  scores   = (float*) alloc((size_t)B_ * P_ * 4);
    float*  alpha    = (float*) alloc((size_t)B_ * P_ * 4);
    float*  context  = (float*) alloc((size_t)B_ * ENC_ * 4);
    float*  gate_pre = (float*) alloc((size_t)B_ * ENC_ * 4);
    float*  x        = (float*) alloc((size_t)B_ * XDIM_ * 4);
    float*  gates    = (float*) alloc((size_t)B_ * G4_ * 4);

    auto gemm = [&](const float* A, const __bf16* BT, const float* bias, float* C,
                    int M, int N, int K, int ldc) {
        int ng = ((N >> 4) + 3) >> 2;          // N tile-groups of 4
        int waves = (M / 16) * ng;
        int blocks = (waves + 3) / 4;
        gemm_bf16_wmma<<<blocks, 128, 0, stream>>>(A, BT, bias, C, M, N, K, ldc);
    };
    auto cblk = [](int total) { return (total + 255) / 256; };

    // ---- one-time: weight conversion to transposed bf16 -------------------
    convT_kernel<<<cblk(ENC_ * ATT_), 256, 0, stream>>>(W_enc_att, WencT, ENC_, ATT_);
    convT_kernel<<<cblk(DEC_ * ATT_), 256, 0, stream>>>(W_dec_att, WdecT, DEC_, ATT_);
    convT_kernel<<<cblk(DEC_ * ENC_), 256, 0, stream>>>(W_fbeta,   WfbT,  DEC_, ENC_);
    convT_kernel<<<cblk(DEC_ * V_),   256, 0, stream>>>(W_fc,      WfcT,  DEC_, V_);
    convT_kernel<<<cblk(ENC_ * DEC_), 256, 0, stream>>>(W_init_h,  WinHT, ENC_, DEC_);
    convT_kernel<<<cblk(ENC_ * DEC_), 256, 0, stream>>>(W_init_c,  WinCT, ENC_, DEC_);
    convT_cat_kernel<<<cblk(G4_ * XDIM_), 256, 0, stream>>>(W_ih, W_hh, WcatT);
    bias_cat_kernel<<<cblk(G4_), 256, 0, stream>>>(b_ih, b_hh, bias_cat);

    // ---- one-time: mean, h0/c0, enc_proj ----------------------------------
    mean_kernel<<<cblk(B_ * ENC_), 256, 0, stream>>>(enc_out, meanb);
    gemm(meanb, WinHT, b_init_h, h, B_, DEC_, ENC_, DEC_);
    gemm(meanb, WinCT, b_init_c, c, B_, DEC_, ENC_, DEC_);
    gemm(enc_out, WencT, b_enc_att, enc_proj, B_ * P_, ATT_, ENC_, ATT_);  // 52 GFLOP

    // ---- sequential decode: 19 steps --------------------------------------
    for (int t = 0; t < T_ - 1; ++t) {
        gemm(h, WdecT, b_dec_att, dec_att, B_, ATT_, DEC_, ATT_);
        scores_kernel<<<(B_ * P_ + 3) / 4, 128, 0, stream>>>(enc_proj, dec_att, w_att, b_att, scores);
        softmax_kernel<<<B_, 256, 0, stream>>>(scores, alpha);
        context_kernel<<<cblk(B_ * ENC_), 256, 0, stream>>>(alpha, enc_out, context);
        gemm(h, WfbT, b_fbeta, gate_pre, B_, ENC_, DEC_, ENC_);
        build_x_kernel<<<cblk(B_ * XDIM_), 256, 0, stream>>>(emb, captions, gate_pre, context, h, x, t);
        gemm(x, WcatT, bias_cat, gates, B_, G4_, XDIM_, G4_);
        lstm_kernel<<<cblk(B_ * DEC_), 256, 0, stream>>>(gates, h, c);
        // logits from updated h, written directly into d_out[b, t, :]
        gemm(h, WfcT, b_fc, out + (size_t)t * V_, B_, V_, DEC_, (T_ - 1) * V_);
    }
}
